// MultiHeadAttention_38706245272176
// MI455X (gfx1250) — matrix operand order repaired and checked by reading, not compiled, not run on hardware
//
#include <hip/hip_runtime.h>
#include <hip/hip_bf16.h>

#define EMB 768
#define HEADS 12
#define HD 64
#define MB 2
#define TT 2048
#define ROWS (MB * TT)   // 4096

typedef __attribute__((ext_vector_type(16))) __bf16 v16bf;
typedef __attribute__((ext_vector_type(8)))  float  v8f;

// A-fragment (16x32 bf16): lane holds row M=lane&15; element e maps to
// K = e + 16*hi, hi = lane>>4
__device__ __forceinline__ int a_koff(int e, int hi) {
    return e + 16 * hi;
}

// ---------------------------------------------------------------------------
// Kernel 1: projection  dst = X @ W^T + b   (X f32 4096x768, W f32 768x768)
// One wave computes a 16x64 tile. store_mode 0: (m,h,t,d)  1: (m,h,d,t)
// ---------------------------------------------------------------------------
__global__ __launch_bounds__(32)
void proj_kernel(const float* __restrict__ X, const float* __restrict__ W,
                 const float* __restrict__ bias, __bf16* __restrict__ dst,
                 int vtrans) {
    const int lane = threadIdx.x;
    const int n  = lane & 15;
    const int hi = lane >> 4;
    const int r0 = blockIdx.x * 16;
    const int c0 = blockIdx.y * 64;

    v8f acc[4] = {v8f{}, v8f{}, v8f{}, v8f{}};

    for (int k0 = 0; k0 < EMB; k0 += 32) {
        v16bf a;
        const float* xr = X + (size_t)(r0 + n) * EMB + k0;
        #pragma unroll
        for (int e = 0; e < 16; ++e) a[e] = (__bf16)xr[a_koff(e, hi)];

        #pragma unroll
        for (int j = 0; j < 4; ++j) {
            v16bf b;
            const float* wr = W + (size_t)(c0 + 16 * j + n) * EMB + k0 + 16 * hi;
            #pragma unroll
            for (int e = 0; e < 16; ++e) b[e] = (__bf16)wr[e];
            acc[j] = __builtin_amdgcn_wmma_f32_16x16x32_bf16(
                false, a, false, b, (short)0, acc[j], false, false);
        }
    }

    #pragma unroll
    for (int j = 0; j < 4; ++j) {
        const int o = c0 + 16 * j + n;
        const float bo = bias[o];
        const int h = o >> 6, d = o & 63;
        #pragma unroll
        for (int i = 0; i < 8; ++i) {
            const int row = r0 + i + 8 * hi;
            const int m = row >> 11, t = row & (TT - 1);
            const float v = acc[j][i] + bo;
            if (vtrans)
                dst[(((size_t)(m * HEADS + h)) * HD + d) * TT + t] = (__bf16)v;
            else
                dst[(((size_t)(m * HEADS + h)) * TT + t) * HD + d] = (__bf16)v;
        }
    }
}

// ---------------------------------------------------------------------------
// Kernel 2: attention. 4 waves (128 threads) cooperate on one (m, h, 16-row
// q tile); the full 16 x 2048 score strip lives in 128KB LDS.
//   phase 1: waves stripe key tiles (QK^T via WMMA)
//   phase 2: each wave softmaxes 4 rows; P written once to d_out
//   phase 3: each wave owns one 16-wide d-tile of ctx = P @ V
// ---------------------------------------------------------------------------
__global__ __launch_bounds__(128)
void attn_kernel(const __bf16* __restrict__ qh, const __bf16* __restrict__ kh,
                 const __bf16* __restrict__ vt, const int* __restrict__ mask,
                 float* __restrict__ attn_out, __bf16* __restrict__ ctxb) {
    __shared__ float sc[16][TT];   // 128 KB

    const int lane = threadIdx.x & 31;
    const int w    = threadIdx.x >> 5;    // wave id 0..3
    const int n  = lane & 15;
    const int hi = lane >> 4;
    const int qr0 = blockIdx.x * 16;
    const int h = blockIdx.y;
    const int m = blockIdx.z;

    const size_t B0 = ((size_t)(m * HEADS + h)) * TT * HD;   // (m,h,t,d)
    const size_t BV = ((size_t)(m * HEADS + h)) * HD * TT;   // (m,h,d,t)
    const float scale = 0.03608439182435161f;                // 1/sqrt(768)

    // q fragments (each wave keeps its own copy; reused across its key tiles)
    v16bf aq0, aq1;
    {
        const __bf16* qr = qh + B0 + (size_t)(qr0 + n) * HD;
        #pragma unroll
        for (int e = 0; e < 16; ++e) {
            aq0[e] = qr[a_koff(e, hi)];
            aq1[e] = qr[32 + a_koff(e, hi)];
        }
    }

    // ---- phase 1: S = scale * q k^T, masked, into LDS (key tiles striped) --
    for (int kt = w; kt < TT / 16; kt += 4) {
        const int key0 = kt * 16;
        v16bf bk0, bk1;
        const __bf16* kr = kh + B0 + (size_t)(key0 + n) * HD;
        #pragma unroll
        for (int e = 0; e < 16; ++e) {
            bk0[e] = kr[e + 16 * hi];
            bk1[e] = kr[32 + e + 16 * hi];
        }
        v8f s = v8f{};
        s = __builtin_amdgcn_wmma_f32_16x16x32_bf16(false, aq1, false, bk1,
                                                    (short)0, s, false, false);
        s = __builtin_amdgcn_wmma_f32_16x16x32_bf16(false, aq0, false, bk0,
                                                    (short)0, s, false, false);
        const int key = key0 + n;
        const int mk = mask[m * TT + key];
        #pragma unroll
        for (int i = 0; i < 8; ++i) {
            float ev = s[i] * scale;
            if (mk == 0) ev = -1e20f;
            sc[i + 8 * hi][key] = ev;
        }
    }
    __syncthreads();

    // ---- phase 2: softmax; wave w owns rows 4w..4w+3; write P once --------
    for (int rr = 0; rr < 4; ++rr) {
        const int r = 4 * w + rr;
        float mx = -3.4e38f;
        for (int c = lane; c < TT; c += 32) mx = fmaxf(mx, sc[r][c]);
        #pragma unroll
        for (int off = 16; off > 0; off >>= 1)
            mx = fmaxf(mx, __shfl_xor(mx, off, 32));

        float sum = 0.f;
        for (int c = lane; c < TT; c += 32) {
            const float ex = __expf(sc[r][c] - mx);
            sc[r][c] = ex;
            sum += ex;
        }
        #pragma unroll
        for (int off = 16; off > 0; off >>= 1)
            sum += __shfl_xor(sum, off, 32);
        const float is = 1.0f / sum;

        float* orow = attn_out + (((size_t)(m * HEADS + h)) * TT + qr0 + r) * TT;
        for (int c = lane; c < TT; c += 32) {
            const float p = sc[r][c] * is;
            sc[r][c] = p;
            orow[c] = p;
        }
    }
    __syncthreads();

    // ---- phase 3: ctx(16x64) = P @ V; wave w owns d-tile dt=w --------------
    v8f acc = v8f{};
    const __bf16* vbase = vt + BV + (size_t)(16 * w + n) * TT;
    for (int kc = 0; kc < TT / 32; ++kc) {
        const int kk0 = kc * 32;
        v16bf ap;
        #pragma unroll
        for (int e = 0; e < 16; ++e) ap[e] = (__bf16)sc[n][kk0 + a_koff(e, hi)];

        v16bf bv;
        const __bf16* vr = vbase + kk0 + 16 * hi;
        #pragma unroll
        for (int e = 0; e < 16; ++e) bv[e] = vr[e];
        acc = __builtin_amdgcn_wmma_f32_16x16x32_bf16(
            false, ap, false, bv, (short)0, acc, false, false);
    }

    const int o = h * HD + 16 * w + n;
    #pragma unroll
    for (int i = 0; i < 8; ++i) {
        const int t = qr0 + i + 8 * hi;
        ctxb[((size_t)(m * TT + t)) * EMB + o] = (__bf16)acc[i];
    }
}

// ---------------------------------------------------------------------------
// Kernel 3: output projection  out = ctx @ Wo^T + bo  (f32 out, row-major)
// ---------------------------------------------------------------------------
__global__ __launch_bounds__(32)
void outproj_kernel(const __bf16* __restrict__ Xb, const float* __restrict__ W,
                    const float* __restrict__ bias, float* __restrict__ out) {
    const int lane = threadIdx.x;
    const int n  = lane & 15;
    const int hi = lane >> 4;
    const int r0 = blockIdx.x * 16;
    const int c0 = blockIdx.y * 64;

    v8f acc[4] = {v8f{}, v8f{}, v8f{}, v8f{}};

    for (int k0 = 0; k0 < EMB; k0 += 32) {
        v16bf a;
        const __bf16* xr = Xb + (size_t)(r0 + n) * EMB + k0;
        #pragma unroll
        for (int e = 0; e < 16; ++e) a[e] = xr[a_koff(e, hi)];

        #pragma unroll
        for (int j = 0; j < 4; ++j) {
            v16bf b;
            const float* wr = W + (size_t)(c0 + 16 * j + n) * EMB + k0 + 16 * hi;
            #pragma unroll
            for (int e = 0; e < 16; ++e) b[e] = (__bf16)wr[e];
            acc[j] = __builtin_amdgcn_wmma_f32_16x16x32_bf16(
                false, a, false, b, (short)0, acc[j], false, false);
        }
    }

    #pragma unroll
    for (int j = 0; j < 4; ++j) {
        const int o = c0 + 16 * j + n;
        const float bo = bias[o];
        #pragma unroll
        for (int i = 0; i < 8; ++i) {
            const int row = r0 + i + 8 * hi;
            out[(size_t)row * EMB + o] = acc[j][i] + bo;
        }
    }
}

// ---------------------------------------------------------------------------
extern "C" void kernel_launch(void* const* d_in, const int* in_sizes, int n_in,
                              void* d_out, int out_size, void* d_ws, size_t ws_size,
                              hipStream_t stream) {
    (void)in_sizes; (void)n_in; (void)out_size; (void)ws_size;

    const float* Q    = (const float*)d_in[0];
    const float* K    = (const float*)d_in[1];
    const float* V    = (const float*)d_in[2];
    const int*   mask = (const int*)  d_in[3];
    const float* Wq   = (const float*)d_in[4];
    const float* bq   = (const float*)d_in[5];
    const float* Wk   = (const float*)d_in[6];
    const float* bk   = (const float*)d_in[7];
    const float* Wv   = (const float*)d_in[8];
    const float* bv   = (const float*)d_in[9];
    const float* Wo   = (const float*)d_in[10];
    const float* bo   = (const float*)d_in[11];

    float* out = (float*)d_out;
    float* attn_out = out + (size_t)MB * TT * EMB;   // tuple: (context, attention)

    const size_t NELEM = (size_t)MB * HEADS * TT * HD;   // 3,145,728
    char* ws = (char*)d_ws;
    __bf16* qh   = (__bf16*)(ws);
    __bf16* kh   = (__bf16*)(ws + 2 * NELEM);
    __bf16* vt   = (__bf16*)(ws + 4 * NELEM);
    __bf16* ctxb = (__bf16*)(ws + 6 * NELEM);

    dim3 gp(ROWS / 16, EMB / 64);      // 256 x 12
    proj_kernel<<<gp, dim3(32), 0, stream>>>(Q, Wq, bq, qh, 0);
    proj_kernel<<<gp, dim3(32), 0, stream>>>(K, Wk, bk, kh, 0);
    proj_kernel<<<gp, dim3(32), 0, stream>>>(V, Wv, bv, vt, 1);

    dim3 ga(TT / 16, HEADS, MB);       // 128 x 12 x 2, 4 waves each
    attn_kernel<<<ga, dim3(128), 0, stream>>>(qh, kh, vt, mask, attn_out, ctxb);

    outproj_kernel<<<gp, dim3(32), 0, stream>>>(ctxb, Wo, bo, out);
}
